// SAABAttention_31516470018719
// MI455X (gfx1250) — compile-verified, matrix operand order checked by reading, not run on hardware
//
#include <hip/hip_runtime.h>
#include <hip/hip_bf16.h>

// ---------------------------------------------------------------------------
// SAAB attention for MI455X (gfx1250, wave32, WMMA 16x16x32 f16).
// D = A(16x32 f16) x B(32x16 f16) + C(16x16 f32)
// Per-lane fragment layout (ISA 7.12.2):
//   A: lane&15 = M row; halfs[0..7]  = K[hi*8 .. hi*8+7]
//                        halfs[8..15] = K[16+hi*8 .. 16+hi*8+7]   (hi = lane>>4)
//   B: same pattern with lane&15 = N col (B stored "N x K" row-major)
//   C/D: vgpr j, lane L: row M = j + 8*(L>>4), col N = L&15
//
// Pipeline:
//   0) f32 -> f16 pre-conversion of activations + weights (one pass, halves
//      all downstream GEMM traffic and removes v_cvt from inner loops)
//   1) QKV projection GEMMs (32x64 per wave, f16 x f16 -> f32)
//   2) flash attention (streaming softmax, LDS P-transpose, prefetch)
//   3) output projection GEMM
// ---------------------------------------------------------------------------

typedef __attribute__((ext_vector_type(16))) _Float16 v16h;
typedef __attribute__((ext_vector_type(8)))  _Float16 v8h;
typedef __attribute__((ext_vector_type(8)))  float    v8f;

#define BATCH   2
#define SLEN    2048
#define DMODEL  1024
#define NHEAD   16
#define HDIM    64
#define MROWS   (BATCH * SLEN)   // 4096
#define SCALE   0.125f           // 1/sqrt(64)

__device__ __forceinline__ v8f vzero8() {
  v8f z;
#pragma unroll
  for (int i = 0; i < 8; ++i) z[i] = 0.0f;
  return z;
}

__device__ __forceinline__ v8f wmma16(v16h a, v16h b, v8f c) {
  return __builtin_amdgcn_wmma_f32_16x16x32_f16(
      /*neg_a=*/false, a, /*neg_b=*/false, b,
      /*c_mod=*/(short)0, c, /*reuse_a=*/false, /*reuse_b=*/false);
}

// Load a 16x32 f16 fragment (A or B) from row-major f16 memory, row stride ld.
__device__ __forceinline__ v16h frag_f16(const _Float16* base, long ld, int lane) {
  const int r  = lane & 15;
  const int hi = (lane >> 4) & 1;
  const _Float16* p = base + (long)r * ld + hi * 8;
  v8h lo = *(const v8h*)(p);
  v8h hh = *(const v8h*)(p + 16);
  v16h f;
#pragma unroll
  for (int i = 0; i < 8; ++i) { f[i] = lo[i]; f[i + 8] = hh[i]; }
  return f;
}

// ---------------------------------------------------------------------------
// Kernel 0: elementwise f32 -> f16 conversion (8 elements / thread, b128 in,
// b128 out).  n must be a multiple of 8 (all our tensors are).
// ---------------------------------------------------------------------------
__global__ __launch_bounds__(256) void cvt_f16_kernel(
    const float* __restrict__ src, _Float16* __restrict__ dst, long n)
{
  const long i = ((long)blockIdx.x * blockDim.x + threadIdx.x) * 8;
  if (i + 8 > n) return;
  float4 a = *(const float4*)(src + i);
  float4 b = *(const float4*)(src + i + 4);
  v8h o;
  o[0]=(_Float16)a.x; o[1]=(_Float16)a.y; o[2]=(_Float16)a.z; o[3]=(_Float16)a.w;
  o[4]=(_Float16)b.x; o[5]=(_Float16)b.y; o[6]=(_Float16)b.z; o[7]=(_Float16)b.w;
  *(v8h*)(dst + i) = o;
}

// ---------------------------------------------------------------------------
// Kernel 1: QKV projection.  Out[m,n] = sum_k X[m,k]*W[n,k] + bias[n]
// blockIdx.z selects {Q,K,V}.  Q,K -> [B,H,S,HD] f16;  V -> [B,H,HD,S] f16.
// One wave computes a 32x64 output tile: 2 A frags + 4 B frags -> 8 WMMAs.
// ---------------------------------------------------------------------------
__global__ __launch_bounds__(128) void qkv_proj_kernel(
    const _Float16* __restrict__ Xf,   // [3][MROWS, D] f16 (q,k,v)
    const _Float16* __restrict__ Wf,   // [3][D, D] f16 (Wq,Wk,Wv)
    const float* __restrict__ bq, const float* __restrict__ bk,
    const float* __restrict__ bv,
    _Float16* __restrict__ Qf, _Float16* __restrict__ Kf, _Float16* __restrict__ Vt)
{
  const int lane = threadIdx.x & 31;
  const int wave = threadIdx.x >> 5;
  const int which = blockIdx.z;

  const _Float16* X    = Xf + (long)which * MROWS * DMODEL;
  const _Float16* W    = Wf + (long)which * DMODEL * DMODEL;
  const float*    bias = (which == 0) ? bq : (which == 1) ? bk : bv;

  const int m0 = (blockIdx.x * 4 + wave) * 32;
  const int n0 = blockIdx.y * 64;

  v8f acc[2][4];
#pragma unroll
  for (int g = 0; g < 2; ++g)
#pragma unroll
    for (int t = 0; t < 4; ++t) acc[g][t] = vzero8();

  for (int k0 = 0; k0 < DMODEL; k0 += 32) {
    v16h a0 = frag_f16(X + (long)(m0)      * DMODEL + k0, DMODEL, lane);
    v16h a1 = frag_f16(X + (long)(m0 + 16) * DMODEL + k0, DMODEL, lane);
#pragma unroll
    for (int t = 0; t < 4; ++t) {
      v16h bfr = frag_f16(W + (long)(n0 + t * 16) * DMODEL + k0, DMODEL, lane);
      acc[0][t] = wmma16(a0, bfr, acc[0][t]);
      acc[1][t] = wmma16(a1, bfr, acc[1][t]);
    }
  }

  const int cn = lane & 15;
  const int hi = (lane >> 4) & 1;
#pragma unroll
  for (int t = 0; t < 4; ++t) {
    const int n  = n0 + t * 16 + cn;
    const int h  = n >> 6;       // /64
    const int hd = n & 63;
    const float bn = bias[n];
#pragma unroll
    for (int g = 0; g < 2; ++g) {
#pragma unroll
      for (int j = 0; j < 8; ++j) {
        const int m = m0 + g * 16 + j + 8 * hi;
        const int b = m >> 11;     // /2048
        const int s = m & (SLEN - 1);
        const _Float16 hv = (_Float16)(acc[g][t][j] + bn);
        if (which == 0)
          Qf[(((long)b * NHEAD + h) * SLEN + s) * HDIM + hd] = hv;
        else if (which == 1)
          Kf[(((long)b * NHEAD + h) * SLEN + s) * HDIM + hd] = hv;
        else
          Vt[(((long)b * NHEAD + h) * HDIM + hd) * SLEN + s] = hv;
      }
    }
  }
}

// ---------------------------------------------------------------------------
// Kernel 2: flash-style attention.  One wave = 16 query rows for one (b,h).
// Streams 32-wide key tiles: 4 score WMMAs, scale + lambda*bias, online
// softmax (shfl_xor row reductions), P transposed through per-wave LDS tile,
// 4 PV WMMAs against V stored [HD, S].  Prefetches next K/V tiles.
// ---------------------------------------------------------------------------
__global__ __launch_bounds__(128) void attn_kernel(
    const _Float16* __restrict__ Qf, const _Float16* __restrict__ Kf,
    const _Float16* __restrict__ Vt,
    const float* __restrict__ bias_struct, const float* __restrict__ lambda_p,
    _Float16* __restrict__ Of)   // [B, S, D] f16
{
  __shared__ __align__(16) _Float16 plds[4][16 * 32];

  const int lane = threadIdx.x & 31;
  const int wave = threadIdx.x >> 5;
  const int q0 = (blockIdx.x * 4 + wave) * 16;
  const int h  = blockIdx.y;
  const int b  = blockIdx.z;
  const float lam = *lambda_p;

  const _Float16* Qb = Qf + ((long)b * NHEAD + h) * SLEN * HDIM;
  const _Float16* Kb = Kf + ((long)b * NHEAD + h) * SLEN * HDIM;
  const _Float16* Vb = Vt + ((long)b * NHEAD + h) * HDIM * SLEN;
  const float*    Bb = bias_struct + ((long)b * SLEN + q0) * SLEN;

  const int cn = lane & 15;
  const int hi = (lane >> 4) & 1;

  // Q fragments (K-dim 0..31 and 32..63), loaded once.
  const v16h aq0 = frag_f16(Qb + (long)q0 * HDIM + 0,  HDIM, lane);
  const v16h aq1 = frag_f16(Qb + (long)q0 * HDIM + 32, HDIM, lane);

  v8f o[4];
#pragma unroll
  for (int t = 0; t < 4; ++t) o[t] = vzero8();
  float mrun[8], lrun[8];
#pragma unroll
  for (int j = 0; j < 8; ++j) { mrun[j] = -3.0e38f; lrun[j] = 0.0f; }

  _Float16* pw = &plds[wave][0];

  for (int k0 = 0; k0 < SLEN; k0 += 32) {
    // prefetch next key/value tiles into cache (global_prefetch_b8)
    {
      const int kn = (k0 + 32 < SLEN) ? (k0 + 32) : k0;
      __builtin_prefetch(Kb + (long)(kn + (lane & 31)) * HDIM, 0, 3);
      __builtin_prefetch(Vb + (long)(lane & 31) * SLEN + kn, 0, 3);
    }

    // scores: two 16x16 tiles (key cols k0..k0+15 and k0+16..k0+31)
    v8f s0 = vzero8(), s1 = vzero8();
    {
      v16h bk0 = frag_f16(Kb + (long)(k0)      * HDIM + 0, HDIM, lane);
      v16h bk1 = frag_f16(Kb + (long)(k0 + 16) * HDIM + 0, HDIM, lane);
      s0 = wmma16(aq0, bk0, s0);
      s1 = wmma16(aq0, bk1, s1);
      bk0 = frag_f16(Kb + (long)(k0)      * HDIM + 32, HDIM, lane);
      bk1 = frag_f16(Kb + (long)(k0 + 16) * HDIM + 32, HDIM, lane);
      s0 = wmma16(aq1, bk0, s0);
      s1 = wmma16(aq1, bk1, s1);
    }

    // scale + structural bias + online softmax update
#pragma unroll
    for (int j = 0; j < 8; ++j) {
      const float* bp = Bb + (long)(j + 8 * hi) * SLEN + k0 + cn;
      float x0 = s0[j] * SCALE + lam * bp[0];
      float x1 = s1[j] * SCALE + lam * bp[16];
      float mv = fmaxf(x0, x1);
#pragma unroll
      for (int msk = 8; msk >= 1; msk >>= 1) mv = fmaxf(mv, __shfl_xor(mv, msk, 32));
      const float mnew = fmaxf(mrun[j], mv);
      const float corr = __expf(mrun[j] - mnew);
      mrun[j] = mnew;
      const float p0 = __expf(x0 - mnew);
      const float p1 = __expf(x1 - mnew);
      float ps = p0 + p1;
#pragma unroll
      for (int msk = 8; msk >= 1; msk >>= 1) ps += __shfl_xor(ps, msk, 32);
      lrun[j] = lrun[j] * corr + ps;
#pragma unroll
      for (int t = 0; t < 4; ++t) o[t][j] *= corr;
      // stash P (C layout -> LDS row-major 16x32)
      pw[(j + 8 * hi) * 32 + cn]      = (_Float16)p0;
      pw[(j + 8 * hi) * 32 + cn + 16] = (_Float16)p1;
    }

    // reload P in A-fragment layout (same-wave LDS store->load is in order)
    v16h ap = frag_f16(pw, 32, lane);

    // out += P (16x32) @ V (32x64), V stored [hd, s] so K-dim is contiguous
#pragma unroll
    for (int t = 0; t < 4; ++t) {
      v16h bvf = frag_f16(Vb + (long)(t * 16) * SLEN + k0, SLEN, lane);
      o[t] = wmma16(ap, bvf, o[t]);
    }
  }

  // normalize and store attention output as f16 [B, S, D]
#pragma unroll
  for (int j = 0; j < 8; ++j) {
    const float inv = 1.0f / lrun[j];
    const int srow = q0 + j + 8 * hi;
    _Float16* op = Of + ((long)b * SLEN + srow) * DMODEL + h * HDIM;
#pragma unroll
    for (int t = 0; t < 4; ++t)
      op[t * 16 + cn] = (_Float16)(o[t][j] * inv);
  }
}

// ---------------------------------------------------------------------------
// Kernel 3: output projection.  Out[m,n] = sum_k A[m,k]*Wo[n,k] + bo[n]  (f32)
// One wave computes a 32x64 output tile.
// ---------------------------------------------------------------------------
__global__ __launch_bounds__(128) void oproj_kernel(
    const _Float16* __restrict__ Af, const _Float16* __restrict__ Wof,
    const float* __restrict__ bo, float* __restrict__ Out)
{
  const int lane = threadIdx.x & 31;
  const int wave = threadIdx.x >> 5;
  const int m0 = (blockIdx.x * 4 + wave) * 32;
  const int n0 = blockIdx.y * 64;

  v8f acc[2][4];
#pragma unroll
  for (int g = 0; g < 2; ++g)
#pragma unroll
    for (int t = 0; t < 4; ++t) acc[g][t] = vzero8();

  for (int k0 = 0; k0 < DMODEL; k0 += 32) {
    v16h a0 = frag_f16(Af + (long)(m0)      * DMODEL + k0, DMODEL, lane);
    v16h a1 = frag_f16(Af + (long)(m0 + 16) * DMODEL + k0, DMODEL, lane);
#pragma unroll
    for (int t = 0; t < 4; ++t) {
      v16h bfr = frag_f16(Wof + (long)(n0 + t * 16) * DMODEL + k0, DMODEL, lane);
      acc[0][t] = wmma16(a0, bfr, acc[0][t]);
      acc[1][t] = wmma16(a1, bfr, acc[1][t]);
    }
  }

  const int cn = lane & 15;
  const int hi = (lane >> 4) & 1;
#pragma unroll
  for (int t = 0; t < 4; ++t) {
    const float bn = bo[n0 + t * 16 + cn];
#pragma unroll
    for (int g = 0; g < 2; ++g)
#pragma unroll
      for (int j = 0; j < 8; ++j)
        Out[(long)(m0 + g * 16 + j + 8 * hi) * DMODEL + n0 + t * 16 + cn] =
            acc[g][t][j] + bn;
  }
}

// ---------------------------------------------------------------------------
extern "C" void kernel_launch(void* const* d_in, const int* in_sizes, int n_in,
                              void* d_out, int out_size, void* d_ws, size_t ws_size,
                              hipStream_t stream) {
  const float* query = (const float*)d_in[0];
  const float* key   = (const float*)d_in[1];
  const float* value = (const float*)d_in[2];
  const float* biasS = (const float*)d_in[3];
  const float* Wq = (const float*)d_in[4];
  const float* bq = (const float*)d_in[5];
  const float* Wk = (const float*)d_in[6];
  const float* bk = (const float*)d_in[7];
  const float* Wv = (const float*)d_in[8];
  const float* bv = (const float*)d_in[9];
  const float* Wo = (const float*)d_in[10];
  const float* bo = (const float*)d_in[11];
  const float* lam = (const float*)d_in[12];
  float* out = (float*)d_out;

  const long XN = (long)MROWS * DMODEL;            // 4M elements per activation
  const long WN = (long)DMODEL * DMODEL;           // 1M elements per weight
  const long HN = (long)BATCH * NHEAD * SLEN * HDIM;  // 4M per head tensor

  _Float16* Xf = (_Float16*)d_ws;        // [3][MROWS, D]   12M halfs
  _Float16* Wf = Xf + 3 * XN;            // [4][D, D]        4M halfs (q,k,v,o)
  _Float16* Qf = Wf + 4 * WN;            // [B,H,S,HD]       4M
  _Float16* Kf = Qf + HN;                // [B,H,S,HD]       4M
  _Float16* Vt = Kf + HN;                // [B,H,HD,S]       4M
  _Float16* Of = Vt + HN;                // [B,S,D]          4M

  dim3 blk128(128);

  // 0) f32 -> f16 conversions
  {
    const int thr = 256;
    cvt_f16_kernel<<<dim3((unsigned)(XN / 8 / thr)), thr, 0, stream>>>(query, Xf + 0 * XN, XN);
    cvt_f16_kernel<<<dim3((unsigned)(XN / 8 / thr)), thr, 0, stream>>>(key,   Xf + 1 * XN, XN);
    cvt_f16_kernel<<<dim3((unsigned)(XN / 8 / thr)), thr, 0, stream>>>(value, Xf + 2 * XN, XN);
    cvt_f16_kernel<<<dim3((unsigned)(WN / 8 / thr)), thr, 0, stream>>>(Wq, Wf + 0 * WN, WN);
    cvt_f16_kernel<<<dim3((unsigned)(WN / 8 / thr)), thr, 0, stream>>>(Wk, Wf + 1 * WN, WN);
    cvt_f16_kernel<<<dim3((unsigned)(WN / 8 / thr)), thr, 0, stream>>>(Wv, Wf + 2 * WN, WN);
    cvt_f16_kernel<<<dim3((unsigned)(WN / 8 / thr)), thr, 0, stream>>>(Wo, Wf + 3 * WN, WN);
  }

  // 1) QKV projections: grid.x = M/128 (4 waves x 32 rows), grid.y = D/64
  qkv_proj_kernel<<<dim3(MROWS / 128, DMODEL / 64, 3), blk128, 0, stream>>>(
      Xf, Wf, bq, bk, bv, Qf, Kf, Vt);

  // 2) attention: grid.x = S/64 (4 q-tiles per block), grid.y = H, grid.z = B
  attn_kernel<<<dim3(SLEN / 64, NHEAD, BATCH), blk128, 0, stream>>>(
      Qf, Kf, Vt, biasS, lam, Of);

  // 3) output projection
  oproj_kernel<<<dim3(MROWS / 128, DMODEL / 64), blk128, 0, stream>>>(
      Of, Wf + 3 * WN, bo, out);
}
